// CachedAttention_67010079752720
// MI455X (gfx1250) — compile-verified, matrix operand order checked by reading, not run on hardware
//
#include <hip/hip_runtime.h>

typedef __attribute__((ext_vector_type(16))) __bf16 v16bf;
typedef __attribute__((ext_vector_type(8)))  __bf16 v8bf;
typedef __attribute__((ext_vector_type(8)))  float  v8f;

#define EMBED   1280
#define NHEADS  32
#define HDIM    40
#define DPAD    64             // head-dim padded for unguarded K-dim fragments
#define DROWS   48             // head-dim padded to 3 n-tiles for AV
#define BSZb    2
#define TGT     1024
#define PASTL   1024
#define SRC     2048           // PASTL + TGT
#define MTOT    2048           // BSZb * TGT
#define SCALING 0.15811388300841897f   // 40^-0.5

// ---------------------------------------------------------------------------
// Unguarded fragment loader for V_WMMA_F32_16X16X32_BF16 A/B operands.
// Row-major tile at `p`, leading dim `ld` (elements), K offset k0.
// Per ISA 16-bit A/B layout each lane holds two contiguous 8-element runs:
//   k = k0 + 8*(lane>>4) + {0..7}  and  + {16..23}
// All call sites are 16-byte aligned (ld in {64,1280,2048}, k0 % 32 == 0),
// so these lower to two b128 loads per fragment.
// ---------------------------------------------------------------------------
__device__ __forceinline__ v16bf frag_fast(const __bf16* p, int ld, int k0) {
  const int lane = threadIdx.x & 31;
  const __bf16* rp = p + (size_t)(lane & 15) * ld + 8 * (lane >> 4) + k0;
  v8bf lo = *(const v8bf*)rp;          // k +0  .. +7
  v8bf hi = *(const v8bf*)(rp + 16);   // k +16 .. +23
  v16bf f;
#pragma unroll
  for (int j = 0; j < 8; ++j) { f[j] = lo[j]; f[8 + j] = hi[j]; }
  return f;
}

// ---------------------------------------------------------------------------
// Prep kernels
// ---------------------------------------------------------------------------
__global__ void k_zero(unsigned int* __restrict__ p, int n) {
  for (int i = blockIdx.x * blockDim.x + threadIdx.x; i < n;
       i += gridDim.x * blockDim.x)
    p[i] = 0u;
}

__global__ void k_cvt_x(const float* __restrict__ x, __bf16* __restrict__ xb, int n) {
  for (int i = blockIdx.x * blockDim.x + threadIdx.x; i < n;
       i += gridDim.x * blockDim.x)
    xb[i] = (__bf16)x[i];
}

__global__ void k_trans_w(const float* __restrict__ w, __bf16* __restrict__ wt) {
  for (int i = blockIdx.x * blockDim.x + threadIdx.x; i < EMBED * EMBED;
       i += gridDim.x * blockDim.x) {
    int k = i / EMBED, n = i % EMBED;
    wt[(size_t)n * EMBED + k] = (__bf16)w[i];   // [K][N] -> [N][K]
  }
}

__global__ void k_past(const float* __restrict__ pk, const float* __restrict__ pv,
                       float* __restrict__ key_out, __bf16* __restrict__ kc,
                       float* __restrict__ val_out, __bf16* __restrict__ vtp) {
  const int total = BSZb * NHEADS * PASTL * HDIM;
  for (int i = blockIdx.x * blockDim.x + threadIdx.x; i < total;
       i += gridDim.x * blockDim.x) {
    int d = i % HDIM;
    int s = (i / HDIM) % PASTL;
    int h = (i / (HDIM * PASTL)) % NHEADS;
    int b = i / (HDIM * PASTL * NHEADS);
    size_t bh = (size_t)(b * NHEADS + h);
    float kv = pk[i], vv = pv[i];
    key_out[(bh * SRC + s) * HDIM + d] = kv;
    kc[(bh * SRC + s) * DPAD + d]      = (__bf16)kv;
    val_out[(bh * SRC + s) * HDIM + d] = vv;
    vtp[(bh * DROWS + d) * SRC + s]    = (__bf16)vv;
  }
}

// ---------------------------------------------------------------------------
// Fused QKV projection.  Register-blocked: each wave owns a 64(M) x 32(N)
// footprint = 4x2 WMMA tiles, 8 accumulators.  Per K-step: 2 B-frags +
// 4 A-frags feed 8 WMMAs.  blockIdx.z selects Q / K / V.
// ---------------------------------------------------------------------------
__global__ void k_qkv(const __bf16* __restrict__ x,
                      const __bf16* __restrict__ wqt, const __bf16* __restrict__ wkt,
                      const __bf16* __restrict__ wvt,
                      const float* __restrict__ bq, const float* __restrict__ bk,
                      const float* __restrict__ bv,
                      __bf16* __restrict__ q_hp,
                      float* __restrict__ key_out, __bf16* __restrict__ kc,
                      float* __restrict__ val_out, __bf16* __restrict__ vtp) {
  const int wave  = threadIdx.x >> 5;
  const int nt0   = (blockIdx.x * 8 + wave) * 2;  // first of 2 n-tiles (0..78)
  const int mt0   = blockIdx.y * 4;               // first of 4 m-tiles (0..124)
  const int which = blockIdx.z;                   // 0=Q 1=K 2=V

  const __bf16* wt   = (which == 0) ? wqt : (which == 1) ? wkt : wvt;
  const float*  bias = (which == 0) ? bq  : (which == 1) ? bk  : bv;

  const __bf16* atile0 = x  + (size_t)mt0 * 16 * EMBED;
  const __bf16* btile0 = wt + (size_t)nt0 * 16 * EMBED;

  v8f acc[4][2] = {};
  for (int k0 = 0; k0 < EMBED; k0 += 32) {
    v16bf b0 = frag_fast(btile0,                      EMBED, k0);
    v16bf b1 = frag_fast(btile0 + (size_t)16 * EMBED, EMBED, k0);
#pragma unroll
    for (int i = 0; i < 4; ++i) {
      v16bf a = frag_fast(atile0 + (size_t)i * 16 * EMBED, EMBED, k0);
      acc[i][0] = __builtin_amdgcn_wmma_f32_16x16x32_bf16(false, a, false, b0,
                                                          (short)0, acc[i][0], false, false);
      acc[i][1] = __builtin_amdgcn_wmma_f32_16x16x32_bf16(false, a, false, b1,
                                                          (short)0, acc[i][1], false, false);
    }
  }

  const int lane = threadIdx.x & 31;
  const int half = lane >> 4;
#pragma unroll
  for (int j = 0; j < 2; ++j) {
    const int n  = (nt0 + j) * 16 + (lane & 15);
    const float bn = bias[n];
    const int h  = n / HDIM, d = n % HDIM;
#pragma unroll
    for (int i = 0; i < 4; ++i) {
#pragma unroll
      for (int r = 0; r < 8; ++r) {
        int m = (mt0 + i) * 16 + r + 8 * half;   // global row in [0,2048)
        float v = acc[i][j][r] + bn;
        int bb = m / TGT, t = m % TGT;
        size_t bh = (size_t)(bb * NHEADS + h);
        if (which == 0) {
          q_hp[(bh * TGT + t) * DPAD + d] = (__bf16)(v * SCALING);
        } else {
          int s = PASTL + t;
          if (which == 1) {
            key_out[(bh * SRC + s) * HDIM + d] = v;
            kc[(bh * SRC + s) * DPAD + d]      = (__bf16)v;
          } else {
            val_out[(bh * SRC + s) * HDIM + d] = v;
            vtp[(bh * DROWS + d) * SRC + s]    = (__bf16)v;
          }
        }
      }
    }
  }
}

// ---------------------------------------------------------------------------
// Attention: one workgroup per (b, h, 16-row Q tile).
// Phase 1: scores[16][2048] (f32, 128KB LDS) via WMMA; Q fragments hoisted,
//          next K tile prefetched (global_prefetch_b8).  All fragment loads
//          unguarded b128 thanks to DPAD=64 zero padding.
// Phase 2: row softmax in LDS, probs -> bf16 strip (64KB LDS)
// Phase 3: AV as 24 tasks (3 d-tiles x 8 s-chunks) across all 8 waves;
//          partials combined via LDS ds_add_f32 into a 16x48 f32 tile.
// ---------------------------------------------------------------------------
__global__ void k_attn(const __bf16* __restrict__ q_hp, const __bf16* __restrict__ kc,
                       const __bf16* __restrict__ vtp, __bf16* __restrict__ attn_bf) {
  extern __shared__ char smem[];
  float*  sc    = (float*)smem;                                        // 16 x 2048 f32
  __bf16* pb    = (__bf16*)(smem + (size_t)16 * SRC * sizeof(float));  // 16 x 2048 bf16
  float*  red   = (float*)(smem + (size_t)16 * SRC * (sizeof(float) + sizeof(__bf16)));
  float*  otile = red + 256;                                           // 16 x 48 f32

  const int mt = blockIdx.x;   // 0..63
  const int h  = blockIdx.y;
  const int b  = blockIdx.z;
  const int wave = threadIdx.x >> 5;
  const int lane = threadIdx.x & 31;
  const int half = lane >> 4;
  const size_t bh = (size_t)(b * NHEADS + h);

  const __bf16* qtile = q_hp + (bh * TGT + mt * 16) * DPAD;
  const __bf16* kbase = kc + bh * SRC * DPAD;

  // zero the AV combine tile (16*48 = 768 floats)
  for (int i = threadIdx.x; i < 16 * 48; i += 256) otile[i] = 0.0f;

  // ---- Phase 1: QK^T ----  (Q fragments are loop-invariant: hoist)
  const v16bf qa0 = frag_fast(qtile, DPAD, 0);
  const v16bf qa1 = frag_fast(qtile, DPAD, 32);
  for (int nt = wave; nt < SRC / 16; nt += 8) {
    if (nt + 8 < SRC / 16)
      __builtin_prefetch(kbase + (size_t)(nt + 8) * 16 * DPAD, 0, 1);
    const __bf16* ktile = kbase + (size_t)nt * 16 * DPAD;
    v16bf b0 = frag_fast(ktile, DPAD, 0);
    v16bf b1 = frag_fast(ktile, DPAD, 32);
    v8f acc = {};
    acc = __builtin_amdgcn_wmma_f32_16x16x32_bf16(false, qa0, false, b0,
                                                  (short)0, acc, false, false);
    acc = __builtin_amdgcn_wmma_f32_16x16x32_bf16(false, qa1, false, b1,
                                                  (short)0, acc, false, false);
#pragma unroll
    for (int r = 0; r < 8; ++r)
      sc[(r + 8 * half) * SRC + nt * 16 + (lane & 15)] = acc[r];
  }
  __syncthreads();

  // ---- Phase 2: softmax (16 rows, 16 threads per row, 128 cols each) ----
  const int row = threadIdx.x >> 4;
  const int sub = threadIdx.x & 15;
  float lmax = -3.0e38f;
  for (int c = sub * 128; c < sub * 128 + 128; ++c)
    lmax = fmaxf(lmax, sc[row * SRC + c]);
  red[threadIdx.x] = lmax;
  __syncthreads();
  float rmax = -3.0e38f;
#pragma unroll
  for (int j = 0; j < 16; ++j) rmax = fmaxf(rmax, red[(row << 4) + j]);
  __syncthreads();

  float lsum = 0.0f;
  for (int c = sub * 128; c < sub * 128 + 128; ++c) {
    float e = __expf(sc[row * SRC + c] - rmax);
    sc[row * SRC + c] = e;
    lsum += e;
  }
  red[threadIdx.x] = lsum;
  __syncthreads();
  float rsum = 0.0f;
#pragma unroll
  for (int j = 0; j < 16; ++j) rsum += red[(row << 4) + j];
  const float inv = 1.0f / rsum;
  for (int c = sub * 128; c < sub * 128 + 128; ++c)
    pb[row * SRC + c] = (__bf16)(sc[row * SRC + c] * inv);
  __syncthreads();

  // ---- Phase 3: AV across all 8 waves ----
  // 24 tasks = 3 d-tiles (d padded 40 -> 48) x 8 s-chunks of 256.
  const __bf16* vbase = vtp + bh * DROWS * SRC;
  for (int task = wave; task < 24; task += 8) {
    const int nt    = task % 3;
    const int chunk = task / 3;
    const __bf16* vtile = vbase + (size_t)nt * 16 * SRC;
    v8f acc = {};
    for (int s0 = chunk * 256; s0 < chunk * 256 + 256; s0 += 32) {
      v16bf a  = frag_fast(pb, SRC, s0);
      v16bf bf = frag_fast(vtile, SRC, s0);
      acc = __builtin_amdgcn_wmma_f32_16x16x32_bf16(false, a, false, bf,
                                                    (short)0, acc, false, false);
    }
    const int n = nt * 16 + (lane & 15);
#pragma unroll
    for (int r = 0; r < 8; ++r)
      atomicAdd(&otile[(r + 8 * half) * 48 + n], acc[r]);   // ds_add_f32
  }
  __syncthreads();

  // write bf16 attention output (16 rows x 40 valid cols)
  for (int i = threadIdx.x; i < 16 * HDIM; i += 256) {
    int m = i / HDIM, d = i % HDIM;
    int t = mt * 16 + m;
    attn_bf[((size_t)(b * TGT + t)) * EMBED + h * HDIM + d] = (__bf16)otile[m * 48 + d];
  }
}

// ---------------------------------------------------------------------------
// Output projection: attn_bf16 @ WoT + bo -> f32 attn_output.
// Same 4x2 register blocking as k_qkv.
// ---------------------------------------------------------------------------
__global__ void k_out(const __bf16* __restrict__ attn_bf, const __bf16* __restrict__ wot,
                      const float* __restrict__ bo, float* __restrict__ out) {
  const int wave = threadIdx.x >> 5;
  const int nt0  = (blockIdx.x * 8 + wave) * 2;
  const int mt0  = blockIdx.y * 4;

  const __bf16* atile0 = attn_bf + (size_t)mt0 * 16 * EMBED;
  const __bf16* btile0 = wot     + (size_t)nt0 * 16 * EMBED;

  v8f acc[4][2] = {};
  for (int k0 = 0; k0 < EMBED; k0 += 32) {
    v16bf b0 = frag_fast(btile0,                      EMBED, k0);
    v16bf b1 = frag_fast(btile0 + (size_t)16 * EMBED, EMBED, k0);
#pragma unroll
    for (int i = 0; i < 4; ++i) {
      v16bf a = frag_fast(atile0 + (size_t)i * 16 * EMBED, EMBED, k0);
      acc[i][0] = __builtin_amdgcn_wmma_f32_16x16x32_bf16(false, a, false, b0,
                                                          (short)0, acc[i][0], false, false);
      acc[i][1] = __builtin_amdgcn_wmma_f32_16x16x32_bf16(false, a, false, b1,
                                                          (short)0, acc[i][1], false, false);
    }
  }

  const int lane = threadIdx.x & 31;
  const int half = lane >> 4;
#pragma unroll
  for (int j = 0; j < 2; ++j) {
    const int n  = (nt0 + j) * 16 + (lane & 15);
    const float bn = bo[n];
#pragma unroll
    for (int i = 0; i < 4; ++i) {
#pragma unroll
      for (int r = 0; r < 8; ++r) {
        int m = (mt0 + i) * 16 + r + 8 * half;
        out[(size_t)m * EMBED + n] = acc[i][j][r] + bn;
      }
    }
  }
}

// ---------------------------------------------------------------------------
extern "C" void kernel_launch(void* const* d_in, const int* in_sizes, int n_in,
                              void* d_out, int out_size, void* d_ws, size_t ws_size,
                              hipStream_t stream) {
  const float* inputs     = (const float*)d_in[0];
  const float* past_key   = (const float*)d_in[1];
  const float* past_value = (const float*)d_in[2];
  const float* Wq = (const float*)d_in[3];
  const float* bq = (const float*)d_in[4];
  const float* Wk = (const float*)d_in[5];
  const float* bk = (const float*)d_in[6];
  const float* Wv = (const float*)d_in[7];
  const float* bv = (const float*)d_in[8];
  const float* Wo = (const float*)d_in[9];
  const float* bo = (const float*)d_in[10];

  // Output layout: attn_output | key_states | value_states (flat, f32)
  float* out_attn = (float*)d_out;
  float* key_out  = out_attn + (size_t)BSZb * TGT * EMBED;
  float* val_out  = key_out  + (size_t)BSZb * NHEADS * SRC * HDIM;

  // Workspace carve-up (bf16 shadows, ~61 MB; all offsets multiples of 16B)
  char* w = (char*)d_ws;
  __bf16* xb      = (__bf16*)w; w += (size_t)MTOT * EMBED * 2;
  __bf16* wqt     = (__bf16*)w; w += (size_t)EMBED * EMBED * 2;
  __bf16* wkt     = (__bf16*)w; w += (size_t)EMBED * EMBED * 2;
  __bf16* wvt     = (__bf16*)w; w += (size_t)EMBED * EMBED * 2;
  __bf16* wot     = (__bf16*)w; w += (size_t)EMBED * EMBED * 2;
  __bf16* q_hp    = (__bf16*)w; w += (size_t)BSZb * NHEADS * TGT * DPAD * 2;
  __bf16* kc      = (__bf16*)w; w += (size_t)BSZb * NHEADS * SRC * DPAD * 2;
  __bf16* vtp     = (__bf16*)w; w += (size_t)BSZb * NHEADS * DROWS * SRC * 2;
  __bf16* attn_bf = (__bf16*)w; w += (size_t)MTOT * EMBED * 2;

  // zero padded shadow buffers (pad lanes must be 0 for unguarded fragments)
  k_zero<<<2048, 256, 0, stream>>>((unsigned int*)q_hp,
                                   (int)((size_t)BSZb * NHEADS * TGT * DPAD / 2));
  k_zero<<<2048, 256, 0, stream>>>((unsigned int*)kc,
                                   (int)((size_t)BSZb * NHEADS * SRC * DPAD / 2));
  k_zero<<<2048, 256, 0, stream>>>((unsigned int*)vtp,
                                   (int)((size_t)BSZb * NHEADS * DROWS * SRC / 2));

  // prep
  k_cvt_x<<<1024, 256, 0, stream>>>(inputs, xb, MTOT * EMBED);
  k_trans_w<<<1024, 256, 0, stream>>>(Wq, wqt);
  k_trans_w<<<1024, 256, 0, stream>>>(Wk, wkt);
  k_trans_w<<<1024, 256, 0, stream>>>(Wv, wvt);
  k_trans_w<<<1024, 256, 0, stream>>>(Wo, wot);
  k_past<<<1024, 256, 0, stream>>>(past_key, past_value, key_out, kc, val_out, vtp);

  // fused QKV projection: 5 x 32 blocks of 8 waves, each wave 64x32 output
  k_qkv<<<dim3(5, 32, 3), 256, 0, stream>>>(xb, wqt, wkt, wvt, bq, bk, bv,
                                            q_hp, key_out, kc, val_out, vtp);

  // attention: ~197 KB dynamic LDS per workgroup (fits CDNA5's 320 KB WGP LDS)
  size_t smem = (size_t)16 * SRC * sizeof(float)     // scores
              + (size_t)16 * SRC * sizeof(__bf16)    // probs
              + 256 * sizeof(float)                  // reduction buffer
              + (size_t)16 * 48 * sizeof(float);     // AV combine tile
  k_attn<<<dim3(TGT / 16, NHEADS, BSZb), 256, smem, stream>>>(q_hp, kc, vtp, attn_bf);

  // output projection
  k_out<<<dim3(5, 32, 1), 256, 0, stream>>>(attn_bf, wot, bo, out_attn);
}
